// BiQRNN_1022202216450
// MI455X (gfx1250) — matrix operand order repaired and checked by reading, not compiled, not run on hardware
//
#include <hip/hip_runtime.h>
#include <hip/hip_bf16.h>

typedef __attribute__((ext_vector_type(16))) _Float16 v16h;
typedef __attribute__((ext_vector_type(8)))  _Float16 v8h;
typedef __attribute__((ext_vector_type(8)))  float    v8f;

#define BATCH   16
#define SEQ     4096
#define EMBD    256
#define HID     256
#define OUTN    64
#define S_CHUNK 128
#define NCHUNK  (SEQ / S_CHUNK)   // 32
#define NCOL    1024              // 2 dirs * (256 Z + 256 F); O-columns skipped entirely

__device__ __forceinline__ float sigm_fast(float x) {
  return 1.0f / (1.0f + __expf(-x));
}
__device__ __forceinline__ float tanh_fast(float x) {
  x = fminf(fmaxf(x, -15.0f), 15.0f);
  float e = __expf(2.0f * x);
  return (e - 1.0f) / (e + 1.0f);
}

__device__ __forceinline__ v16h cat8(v8h lo, v8h hi) {
  return __builtin_shufflevector(lo, hi, 0, 1, 2, 3, 4, 5, 6, 7,
                                         8, 9, 10, 11, 12, 13, 14, 15);
}

// ---------------------------------------------------------------------------
// Kernel 0: convert the used 2*512 weight rows fp32 -> f16, layout [c][k],
//           c = dir*512 + zf*256 + h   (zf: 0=Z rows 0..255, 1=F rows 256..511)
// ---------------------------------------------------------------------------
__global__ void wconv_kernel(const float* __restrict__ wf,
                             const float* __restrict__ wb,
                             _Float16* __restrict__ whalf) {
  int t = blockIdx.x * blockDim.x + threadIdx.x;   // 0 .. NCOL*EMBD-1
  int c = t >> 8;
  int k = t & 255;
  int dir = c >> 9;
  int zf  = (c >> 8) & 1;
  int h   = c & 255;
  const float* src = dir ? wb : wf;
  whalf[t] = (_Float16)src[(zf * 256 + h) * EMBD + k];
}

// ---------------------------------------------------------------------------
// Kernel 1: per (batch, s-chunk) block (256 thr = 8 waves):
//   gather emb rows -> f16 LDS tile (XOR-swizzled 16B granules),
//   per wave: 4 channel-groups OUTER (B-fragments register-resident),
//   time-tiles INNER; WMMA f16 GEMM for Z and F, fused tanh/sigmoid +
//   affine scan composition; emit per-chunk (a,b).
// ---------------------------------------------------------------------------
__global__ __launch_bounds__(256, 1)
void qrnn_chunk_kernel(const int*   __restrict__ X,
                       const float* __restrict__ emb,
                       const float* __restrict__ bf,
                       const float* __restrict__ bb,
                       const _Float16* __restrict__ whalf,
                       float2* __restrict__ chunkAB) {
  // LDS A tile: 128 rows x 256 f16; granule g (8 halves) of row r lives at
  // (g ^ (r & 7)) -> conflict-free ds_load_b128 fragment reads. 64 KB.
  __shared__ _Float16 As[S_CHUNK * EMBD];

  const int b     = blockIdx.x / NCHUNK;
  const int chunk = blockIdx.x % NCHUNK;
  const int s0    = chunk * S_CHUNK;
  const int tid   = threadIdx.x;

  // ---- phase 1: gather + fp32->f16 convert into LDS (2 threads per row) ----
  {
    const int r    = tid >> 1;          // 0..127
    const int half = tid & 1;           // which 128-float half of the row
    const int tok  = X[b * SEQ + s0 + r];
    const float* src = emb + (size_t)tok * EMBD + half * 128;
    _Float16* rowb = &As[r * EMBD];
    const int sw = r & 7;
#pragma unroll
    for (int j = 0; j < 16; ++j) {      // 16 granules of 8 halves
      float4 x0 = *(const float4*)(src + j * 8);
      float4 x1 = *(const float4*)(src + j * 8 + 4);
      v8h hv;
      hv[0] = (_Float16)x0.x; hv[1] = (_Float16)x0.y;
      hv[2] = (_Float16)x0.z; hv[3] = (_Float16)x0.w;
      hv[4] = (_Float16)x1.x; hv[5] = (_Float16)x1.y;
      hv[6] = (_Float16)x1.z; hv[7] = (_Float16)x1.w;
      int g = half * 16 + j;
      *(v8h*)(rowb + ((g ^ sw) << 3)) = hv;
    }
  }
  __syncthreads();

  const int wave  = tid >> 5;
  const int lane  = tid & 31;
  const int m     = lane & 15;     // row (A) / column (B,C,D) within tile
  const int khalf = lane >> 4;
  const int sw    = m & 7;

  // channel-group OUTER loop (rolled): B fragments stay register-resident
#pragma unroll 1
  for (int g = 0; g < 4; ++g) {
    const int grp = wave * 4 + g;    // 0..31
    const int dir = grp >> 4;        // 0=fwd, 1=bwd
    const int ht  = grp & 15;        // 16-channel tile within H
    const bool bwd = (dir != 0);
    const int colZ = dir * 512 + ht * 16 + m;   // this lane's Z column
    const int colF = colZ + 256;                // this lane's F column

    // ---- load B fragments ONCE per group: 16 x v16h = 128 VGPRs ----
    v16h bzf[8], bgf[8];
    {
      const _Float16* wz = whalf + colZ * EMBD + khalf * 16;
      const _Float16* wg = whalf + colF * EMBD + khalf * 16;
#pragma unroll
      for (int k = 0; k < 8; ++k) {
        v8h zl = *(const v8h*)(wz + k * 32);
        v8h zh = *(const v8h*)(wz + k * 32 + 8);
        v8h gl = *(const v8h*)(wg + k * 32);
        v8h gh = *(const v8h*)(wg + k * 32 + 8);
        bzf[k] = cat8(zl, zh);
        bgf[k] = cat8(gl, gh);
      }
    }

    const int   hch = ht * 16 + m;
    const float bzs = (dir ? bb : bf)[hch];
    const float bfs = (dir ? bb : bf)[256 + hch];

    float run_a = 1.0f, run_b = 0.0f;

#pragma unroll 1
    for (int tt = 0; tt < S_CHUNK / 16; ++tt) {
      const _Float16* arow = &As[(tt * 16 + m) * EMBD];
      v8f accz = {}, accf = {};
#pragma unroll
      for (int k = 0; k < 8; ++k) {
        int gl = k * 4 + khalf * 2;    // granule of K = k*32 + khalf*16
        int gh = gl + 1;               // granule of K = k*32 + khalf*16 + 8
        v8h lo = *(const v8h*)(arow + ((gl ^ sw) << 3));
        v8h hi = *(const v8h*)(arow + ((gh ^ sw) << 3));
        v16h afr = cat8(lo, hi);
        accz = __builtin_amdgcn_wmma_f32_16x16x32_f16(false, afr, false, bzf[k],
                                                      (short)0, accz, false, false);
        accf = __builtin_amdgcn_wmma_f32_16x16x32_f16(false, afr, false, bgf[k],
                                                      (short)0, accf, false, false);
      }

      // activations: lane holds times (tt*16 + khalf*8 + v) of channel hch
      float zv[8], fv[8];
#pragma unroll
      for (int v = 0; v < 8; ++v) {
        zv[v] = tanh_fast(accz[v] + bzs);
        fv[v] = sigm_fast(accf[v] + bfs);
      }
      // in-lane affine composition (time order depends on direction)
      float la = 1.0f, lb = 0.0f;
      if (!bwd) {
#pragma unroll
        for (int v = 0; v < 8; ++v) { lb = fv[v] * lb + (1.0f - fv[v]) * zv[v]; la *= fv[v]; }
      } else {
#pragma unroll
        for (int v = 7; v >= 0; --v) { lb = fv[v] * lb + (1.0f - fv[v]) * zv[v]; la *= fv[v]; }
      }
      // combine the two time-halves of this 16-step tile (cross-lane)
      float a_lo = __shfl(la, m, 32),      b_lo = __shfl(lb, m, 32);
      float a_hi = __shfl(la, m + 16, 32), b_hi = __shfl(lb, m + 16, 32);
      float ta, tb;
      if (!bwd) { ta = a_hi * a_lo; tb = a_hi * b_lo + b_hi; }   // hi after lo
      else      { ta = a_lo * a_hi; tb = a_lo * b_hi + b_lo; }   // lo after hi
      // fold tile into running chunk composite
      if (!bwd) { run_b = ta * run_b + tb; run_a *= ta; }        // tile applied after
      else      { run_b = run_a * tb + run_b; run_a *= ta; }     // tile applied before
    }

    if (lane < 16) {
      const int hw = ht * 16 + lane;
      size_t idx = ((((size_t)dir * BATCH + b) * HID) + hw) * NCHUNK + chunk;
      chunkAB[idx] = make_float2(run_a, run_b);
    }
  }
}

// ---------------------------------------------------------------------------
// Kernel 2: fold the 32 chunk-affines per (dir, b, h); h_final = b_total.
// ---------------------------------------------------------------------------
__global__ void qrnn_combine_kernel(const float2* __restrict__ chunkAB,
                                    float* __restrict__ hbuf) {
  int t = blockIdx.x * blockDim.x + threadIdx.x;   // 0 .. 2*16*256-1
  if (t >= 2 * BATCH * HID) return;
  int dir = t / (BATCH * HID);
  int rem = t % (BATCH * HID);
  int b = rem / HID;
  int h = rem % HID;
  const float2* c = chunkAB + (size_t)t * NCHUNK;  // layout matches t ordering
  float ra = 1.0f, rb = 0.0f;
  if (dir == 0) {
    for (int ch = 0; ch < NCHUNK; ++ch) { float2 v = c[ch]; rb = v.x * rb + v.y; ra *= v.x; }
  } else {
    for (int ch = NCHUNK - 1; ch >= 0; --ch) { float2 v = c[ch]; rb = v.x * rb + v.y; ra *= v.x; }
  }
  hbuf[b * (2 * HID) + dir * HID + h] = rb;
}

// ---------------------------------------------------------------------------
// Kernel 3: out[b][o] = b_out[o] + sum_j h[b][j] * w_out[o][j]   (16x64, K=512)
// ---------------------------------------------------------------------------
__global__ void qrnn_out_kernel(const float* __restrict__ hbuf,
                                const float* __restrict__ w_out,
                                const float* __restrict__ b_out,
                                float* __restrict__ out) {
  int b = blockIdx.x;
  int o = threadIdx.x;
  const float* hv = hbuf + b * (2 * HID);
  const float* wv = w_out + o * (2 * HID);
  float acc = b_out[o];
#pragma unroll 8
  for (int j = 0; j < 2 * HID; ++j) acc = fmaf(hv[j], wv[j], acc);
  out[b * OUTN + o] = acc;
}

// ---------------------------------------------------------------------------
extern "C" void kernel_launch(void* const* d_in, const int* in_sizes, int n_in,
                              void* d_out, int out_size, void* d_ws, size_t ws_size,
                              hipStream_t stream) {
  const int*   X    = (const int*)d_in[0];
  const float* emb  = (const float*)d_in[1];
  const float* wf   = (const float*)d_in[2];
  const float* bf   = (const float*)d_in[3];
  const float* wb   = (const float*)d_in[4];
  const float* bb   = (const float*)d_in[5];
  const float* wout = (const float*)d_in[6];
  const float* bout = (const float*)d_in[7];
  float* out = (float*)d_out;

  char* ws = (char*)d_ws;
  _Float16* whalf   = (_Float16*)ws;                                   // 512 KB
  size_t off = (size_t)NCOL * EMBD * sizeof(_Float16);
  float2*   chunkAB = (float2*)(ws + off);                             // 2 MB
  off += (size_t)2 * BATCH * HID * NCHUNK * sizeof(float2);
  float*    hbuf    = (float*)(ws + off);                              // 32 KB

  wconv_kernel<<<(NCOL * EMBD) / 256, 256, 0, stream>>>(wf, wb, whalf);
  qrnn_chunk_kernel<<<BATCH * NCHUNK, 256, 0, stream>>>(X, emb, bf, bb, whalf, chunkAB);
  qrnn_combine_kernel<<<(2 * BATCH * HID) / 256, 256, 0, stream>>>(chunkAB, hbuf);
  qrnn_out_kernel<<<BATCH, OUTN, 0, stream>>>(hbuf, wout, bout, out);
}
